// SubgraphMOE_42863773614347
// MI455X (gfx1250) — matrix operand order repaired and checked by reading, not compiled, not run on hardware
//
#include <hip/hip_runtime.h>
#include <hip/hip_bf16.h>

// ---------------------------------------------------------------------------
// SubgraphMOE on gfx1250 (MI455X): WMMA bf16 16x16x32 pipeline, round 3.
//   B=8 N=200 T=12 D=64 C=4 E=8 P=32 TOPK=2, S=1000, tokens=96000.
// prep kernel: f32 weights -> bf16 B-operand lane layout in d_ws.
// main kernel: each wave processes TWO 16-token tiles per iteration:
//   - every W/K/M B-operand register load feeds two independent WMMA chains
//     (halves streamed weight traffic, doubles ILP),
//   - router math is full-wave (lanes 0-15 tile0, lanes 16-31 tile1),
//   - attention softmax is done in registers on the C-layout accumulators
//     with __shfl_xor 16-lane reductions (full-wave v_exp, no f32 staging),
//   - gate weights folded into aw rows so one WMMA accumulator per tile sums
//     all 16 expert-branch contributions; residual re-reads x from L2.
// ---------------------------------------------------------------------------

typedef __attribute__((ext_vector_type(16))) __bf16 v16bf;
typedef __attribute__((ext_vector_type(8)))  __bf16 v8bf;
typedef __attribute__((ext_vector_type(2)))  __bf16 v2bf;
typedef __attribute__((ext_vector_type(8)))  float  v8f;

#define Bdim 8
#define Ndim 200
#define Tdim 12
#define Ddim 64
#define Cdim 4
#define Sdim (Ndim + Cdim * Ndim)          // 1000
#define NTOK (Bdim * Sdim * Tdim)          // 96000
#define NTILES (NTOK / 16)                 // 6000
#define NPAIRS (NTILES / 2)                // 3000
#define MAINSZ ((size_t)Bdim * Ndim * Tdim * Ddim)   // 1228800

// d_ws bf16 element offsets
#define WS_WB 0                   // [16 eb][2 kblk][4 ntile][32 lane][16]  = 65536
#define WS_KB 65536               // [16 eb][2 kblk][2 ptile][32 lane][16]  = 32768
#define WS_MB 98304               // [16 eb][4 ntile][32 lane][16]          = 32768
#define WS_RB 131072              // [2 kblk][32 lane][16]                  = 1024
#define WS_TOTAL 132096

// LDS byte offsets
#define L_WB    0                 // 131072 B  (WB copied from d_ws)
#define L_RB    131072            // 2048 B
#define L_BIAS  133120            // 4096 B  [2][8][64] f32
#define L_RC    137216            // 96 B    br|gamma|beta f32
#define L_WAVE  137312            // per-wave scratch
// per-wave scratch (bytes from wave base), two tile slots (suffix 0/1):
//   XB0  0      XB1  2304   : bf16 16x72 x tiles (row-major)
//   QB0  4608   QB1  6912   : bf16 16x72 q tiles (also f32 router staging)
//   AWB0 9216   AWB1 10496  : bf16 16x40 g*softmax(aw) (row-major)
//   G0   11776  G1   12288  : f32 16x8 gates
//   GT0  12800  GT1  12864  : f32 16 gate totals
//   OB0  12928  OB1  13056  : u64 16 output bases
#define WAVE_BYTES 13184
#define WAVES 2
#define SMEM_BYTES (L_WAVE + WAVES * WAVE_BYTES)   // 163680 <= 163840

__device__ __forceinline__ void wave_lds_fence() {
  asm volatile("s_wait_dscnt 0" ::: "memory");
}

__device__ __forceinline__ v8f wmma_bf16(v16bf a, v16bf b, v8f c) {
  return __builtin_amdgcn_wmma_f32_16x16x32_bf16(false, a, false, b,
                                                 (short)0, c, false, false);
}

// A-operand (16-bit 16x32): lane holds row, K = {koff..koff+15}
// -> two adjacent 16B runs in a row-major bf16 row.
__device__ __forceinline__ v16bf load_a(const __bf16* rowbase, int koff) {
  v8bf lo = *(const v8bf*)(rowbase + koff);
  v8bf hi = *(const v8bf*)(rowbase + koff + 8);
  return __builtin_shufflevector(lo, hi, 0, 1, 2, 3, 4, 5, 6, 7,
                                 8, 9, 10, 11, 12, 13, 14, 15);
}

__device__ __forceinline__ float fast_tanh(float x) {
  x = fminf(fmaxf(x, -15.f), 15.f);
  float e = __expf(2.f * x);
  return (e - 1.f) / (e + 1.f);
}

// ------------------------- weight prep (f32 -> bf16 B-operand layout) ------
__global__ void moe_prep(const float* __restrict__ Ws, const float* __restrict__ Wt,
                         const float* __restrict__ Ks, const float* __restrict__ Kt,
                         const float* __restrict__ Ms, const float* __restrict__ Mt,
                         const float* __restrict__ Wr, __bf16* __restrict__ wsb) {
  int idx = blockIdx.x * blockDim.x + threadIdx.x;
  if (idx >= WS_TOTAL) return;
  float val = 0.f;
  if (idx < WS_KB) {                       // WB: B[k][n] = W[e][n][k]
    int i = idx & 15, lane = (idx >> 4) & 31, nt = (idx >> 9) & 3;
    int kb = (idx >> 11) & 1, eb = idx >> 12;
    int n = nt * 16 + (lane & 15);
    int k = kb * 32 + ((lane >> 4) << 4) + i;
    const float* W = (eb >> 3) ? Wt : Ws;
    val = W[((eb & 7) * 64 + n) * 64 + k];
  } else if (idx < WS_MB) {                // KB: B[k][p] = K[e][p][k]
    int r = idx - WS_KB;
    int i = r & 15, lane = (r >> 4) & 31, pt = (r >> 9) & 1;
    int kb = (r >> 10) & 1, eb = r >> 11;
    int p = pt * 16 + (lane & 15);
    int k = kb * 32 + ((lane >> 4) << 4) + i;
    const float* K = (eb >> 3) ? Kt : Ks;
    val = K[((eb & 7) * 32 + p) * 64 + k];
  } else if (idx < WS_RB) {                // MB: B[p][n] = M[e][p][n]
    int r = idx - WS_MB;
    int i = r & 15, lane = (r >> 4) & 31, nt = (r >> 9) & 3, eb = r >> 11;
    int n = nt * 16 + (lane & 15);
    int p = ((lane >> 4) << 4) + i;
    const float* M = (eb >> 3) ? Mt : Ms;
    val = M[((eb & 7) * 32 + p) * 64 + n];
  } else {                                 // RB: B[k][e] = Wr[e][k], cols 8..15 = 0
    int r = idx - WS_RB;
    int i = r & 15, lane = (r >> 4) & 31, kb = r >> 9;
    int n = lane & 15;
    int k = kb * 32 + ((lane >> 4) << 4) + i;
    val = (n < 8) ? Wr[n * 64 + k] : 0.f;
  }
  wsb[idx] = (__bf16)val;
}

// ------------------------------- main kernel -------------------------------
__global__ void __launch_bounds__(32 * WAVES, 1)
moe_main(const float* __restrict__ mg, const float* __restrict__ sg,
         const float* __restrict__ brr, const float* __restrict__ gam,
         const float* __restrict__ bet, const float* __restrict__ bsv,
         const float* __restrict__ btv, const __bf16* __restrict__ wsb,
         float* __restrict__ out) {
  extern __shared__ char smem[];
  const int tid = threadIdx.x;
  const int lane = tid & 31, wave = tid >> 5;

  // --- prologue: stage W-operands + router consts + biases into LDS ---
  {
    uint4* dWB = (uint4*)(smem + L_WB);
    const uint4* sW = (const uint4*)(wsb + WS_WB);
    for (int i = tid; i < 131072 / 16; i += blockDim.x) dWB[i] = sW[i];
    uint4* dRB = (uint4*)(smem + L_RB);
    const uint4* sR = (const uint4*)(wsb + WS_RB);
    for (int i = tid; i < 2048 / 16; i += blockDim.x) dRB[i] = sR[i];
    float* sBias = (float*)(smem + L_BIAS);
    for (int i = tid; i < 512; i += blockDim.x) {
      sBias[i] = bsv[i];
      sBias[512 + i] = btv[i];
    }
    float* sRC = (float*)(smem + L_RC);
    if (tid < 8) { sRC[tid] = brr[tid]; sRC[8 + tid] = gam[tid]; sRC[16 + tid] = bet[tid]; }
  }
  __syncthreads();

  const __bf16* sWB = (const __bf16*)(smem + L_WB);
  const __bf16* sRB = (const __bf16*)(smem + L_RB);
  const float*  sBias = (const float*)(smem + L_BIAS);
  const float*  sRC = (const float*)(smem + L_RC);
  char* swb = smem + L_WAVE + wave * WAVE_BYTES;
  __bf16* sXB[2]  = {(__bf16*)(swb + 0),     (__bf16*)(swb + 2304)};
  __bf16* sQB[2]  = {(__bf16*)(swb + 4608),  (__bf16*)(swb + 6912)};
  float*  sLQ[2]  = {(float*)(swb + 4608),   (float*)(swb + 6912)};  // router stage
  __bf16* sAWB[2] = {(__bf16*)(swb + 9216),  (__bf16*)(swb + 10496)};
  float*  sG[2]   = {(float*)(swb + 11776),  (float*)(swb + 12288)};
  float*  sGT[2]  = {(float*)(swb + 12800),  (float*)(swb + 12864)};
  size_t* sOB[2]  = {(size_t*)(swb + 12928), (size_t*)(swb + 13056)};

  const __bf16* wsKB = wsb + WS_KB;
  const __bf16* wsMB = wsb + WS_MB;

  const int mrow = lane & 15;
  const int koff = (lane >> 4) << 4;        // A-layout K offset (0 / 16)
  const int hi8 = (lane >> 4) << 3;         // C-layout row offset
  const int cl = lane & 15;                 // C-layout column

  for (int pair = blockIdx.x * WAVES + wave; pair < NPAIRS;
       pair += gridDim.x * WAVES) {
    // ---- load 2 x 16 token rows of x, convert to bf16 row-major in LDS ----
#pragma unroll
    for (int u = 0; u < 2; ++u) {
      int tile = pair * 2 + u;
      int row = lane & 15, half = lane >> 4;
      int flat = tile * 16 + row;
      int tt = flat % Tdim;
      int bs_ = flat / Tdim;
      int s = bs_ % Sdim, b = bs_ / Sdim;
      const float* src;
      int stride;
      size_t obase;
      if (s < Ndim) {
        size_t base = (((size_t)b * Ndim + s) * Tdim + tt) * Ddim;
        src = mg + base;  stride = 1;  obase = base;
      } else {
        int sp = s - Ndim;
        int c = sp / Ndim, ni = sp % Ndim;
        size_t base = (((size_t)b * Ndim + ni) * Tdim + tt) * Ddim;
        src = sg + base * Cdim + c;  stride = Cdim;
        obase = MAINSZ + base * Cdim + c;
      }
      __bf16* dst = sXB[u] + row * 72 + half * 32;
#pragma unroll
      for (int j = 0; j < 32; j += 2) {
        float a = src[(half * 32 + j) * stride];
        float b2 = src[(half * 32 + j + 1) * stride];
        *(v2bf*)(dst + j) = v2bf{(__bf16)a, (__bf16)b2};
      }
      if (lane < 16) sOB[u][row] = obase;
    }
    wave_lds_fence();

    // ---- X as WMMA A-operands for both tiles ----
    v16bf xa0[2], xa1[2];
#pragma unroll
    for (int u = 0; u < 2; ++u) {
      xa0[u] = load_a(sXB[u] + mrow * 72, koff);
      xa1[u] = load_a(sXB[u] + mrow * 72 + 32, koff);
    }

    // ---- router logits via WMMA, staged into (unused) Q buffers as f32 ----
    {
      v16bf rb0 = *(const v16bf*)(sRB + lane * 16);
      v16bf rb1 = *(const v16bf*)(sRB + 512 + lane * 16);
#pragma unroll
      for (int u = 0; u < 2; ++u) {
        v8f rl = {};
        rl = wmma_bf16(xa0[u], rb0, rl);
        rl = wmma_bf16(xa1[u], rb1, rl);
#pragma unroll
        for (int v = 0; v < 8; ++v) sLQ[u][(v + hi8) * 16 + cl] = rl[v];
      }
    }
    wave_lds_fence();
    {  // full-wave router: lanes 0-15 -> tile0 tokens, 16-31 -> tile1 tokens
      int u = lane >> 4, tok = lane & 15;
      const float* lq = sLQ[u];
      float lg[8];
#pragma unroll
      for (int e = 0; e < 8; ++e) lg[e] = lq[tok * 16 + e] + sRC[e];
      float mu = 0.f;
#pragma unroll
      for (int e = 0; e < 8; ++e) mu += lg[e];
      mu *= 0.125f;
      float var = 0.f;
#pragma unroll
      for (int e = 0; e < 8; ++e) { float d = lg[e] - mu; var += d * d; }
      var *= 0.125f;
      float rs = rsqrtf(var + 1e-5f);
      float p[8], mx = -1e30f;
#pragma unroll
      for (int e = 0; e < 8; ++e) {
        p[e] = (lg[e] - mu) * rs * sRC[8 + e] + sRC[16 + e];
        mx = fmaxf(mx, p[e]);
      }
      float sum = 0.f;
#pragma unroll
      for (int e = 0; e < 8; ++e) { p[e] = __expf(p[e] - mx); sum += p[e]; }
      float inv = 1.f / sum;
#pragma unroll
      for (int e = 0; e < 8; ++e) p[e] *= inv;
      int i0 = 0;
      for (int e = 1; e < 8; ++e) if (p[e] > p[i0]) i0 = e;
      int i1 = (i0 == 0) ? 1 : 0;
      for (int e = 0; e < 8; ++e) if (e != i0 && p[e] > p[i1]) i1 = e;
      float ssum = p[i0] + p[i1];
      float ginv = 1.f / (ssum + 1e-8f);
#pragma unroll
      for (int e = 0; e < 8; ++e)
        sG[u][tok * 8 + e] = (e == i0 || e == i1) ? p[e] * ginv : 0.f;
      sGT[u][tok] = ssum * ginv;
    }
    wave_lds_fence();

    // ---- expert/branch loop over both tiles ----
    v8f acc[2][4] = {{v8f{}, v8f{}, v8f{}, v8f{}}, {v8f{}, v8f{}, v8f{}, v8f{}}};
    for (int eb = 0; eb < 16; ++eb) {         // eb = branch*8 + expert
      const float* bias = sBias + eb * 64;
      // Q = tanh(X W^T + b) for both tiles; each wb pair feeds 2 chains
#pragma unroll
      for (int nt = 0; nt < 4; ++nt) {
        v16bf wb0 = *(const v16bf*)(sWB + (((eb * 2 + 0) * 4 + nt) * 32 + lane) * 16);
        v16bf wb1 = *(const v16bf*)(sWB + (((eb * 2 + 1) * 4 + nt) * 32 + lane) * 16);
        float bv = bias[nt * 16 + cl];
#pragma unroll
        for (int u = 0; u < 2; ++u) {
          v8f qc = {bv, bv, bv, bv, bv, bv, bv, bv};
          qc = wmma_bf16(xa0[u], wb0, qc);
          qc = wmma_bf16(xa1[u], wb1, qc);
#pragma unroll
          for (int v = 0; v < 8; ++v)
            sQB[u][(v + hi8) * 72 + nt * 16 + cl] = (__bf16)fast_tanh(qc[v]);
        }
      }
      wave_lds_fence();
      // aw logits 16x32 = Q K^T, both tiles; softmax+gate in registers
      v16bf qa0[2], qa1[2];
#pragma unroll
      for (int u = 0; u < 2; ++u) {
        qa0[u] = load_a(sQB[u] + mrow * 72, koff);
        qa1[u] = load_a(sQB[u] + mrow * 72 + 32, koff);
      }
      v8f ac[2][2];
#pragma unroll
      for (int pt = 0; pt < 2; ++pt) {
        v16bf kb0 = *(const v16bf*)(wsKB + (((eb * 2 + 0) * 2 + pt) * 32 + lane) * 16);
        v16bf kb1 = *(const v16bf*)(wsKB + (((eb * 2 + 1) * 2 + pt) * 32 + lane) * 16);
#pragma unroll
        for (int u = 0; u < 2; ++u) {
          v8f a = {};
          a = wmma_bf16(qa0[u], kb0, a);
          a = wmma_bf16(qa1[u], kb1, a);
          ac[u][pt] = a;
        }
      }
      // register softmax: per row (vgpr) reduce across the 16-lane half
#pragma unroll
      for (int u = 0; u < 2; ++u) {
#pragma unroll
        for (int v = 0; v < 8; ++v) {
          float m = fmaxf(ac[u][0][v], ac[u][1][v]);
          m = fmaxf(m, __shfl_xor(m, 1, 32));
          m = fmaxf(m, __shfl_xor(m, 2, 32));
          m = fmaxf(m, __shfl_xor(m, 4, 32));
          m = fmaxf(m, __shfl_xor(m, 8, 32));
          float e0 = __expf(ac[u][0][v] - m);
          float e1 = __expf(ac[u][1][v] - m);
          float s = e0 + e1;
          s += __shfl_xor(s, 1, 32);
          s += __shfl_xor(s, 2, 32);
          s += __shfl_xor(s, 4, 32);
          s += __shfl_xor(s, 8, 32);
          float g = sG[u][(v + hi8) * 8 + (eb & 7)];   // broadcast read
          float sc = g / s;
          sAWB[u][(v + hi8) * 40 + cl]      = (__bf16)(e0 * sc);
          sAWB[u][(v + hi8) * 40 + 16 + cl] = (__bf16)(e1 * sc);
        }
      }
      wave_lds_fence();
      // (g * aw) @ M accumulated into per-tile running C; mb feeds 2 chains
      v16bf awa[2];
#pragma unroll
      for (int u = 0; u < 2; ++u) awa[u] = load_a(sAWB[u] + mrow * 40, koff);
#pragma unroll
      for (int nt = 0; nt < 4; ++nt) {
        v16bf mb = *(const v16bf*)(wsMB + ((eb * 4 + nt) * 32 + lane) * 16);
#pragma unroll
        for (int u = 0; u < 2; ++u)
          acc[u][nt] = wmma_bf16(awa[u], mb, acc[u][nt]);
      }
      wave_lds_fence();
    }

    // ---- residual g_tot * x (re-read from L2-resident input), scatter ----
#pragma unroll
    for (int u = 0; u < 2; ++u) {
#pragma unroll
      for (int v = 0; v < 8; ++v) {
        int r = v + hi8;
        size_t ob = sOB[u][r];
        float gt = sGT[u][r];
        const float* xp;
        size_t xi;
        int st;
        if (ob < MAINSZ) { xp = mg; xi = ob; st = 1; }
        else             { xp = sg; xi = ob - MAINSZ; st = Cdim; }
#pragma unroll
        for (int nt = 0; nt < 4; ++nt) {
          int col = nt * 16 + cl;
          float x = xp[xi + (size_t)col * st];
          out[ob + (size_t)col * st] = acc[u][nt][v] + gt * x;
        }
      }
    }
  }
}

// ------------------------------- launcher ----------------------------------
extern "C" void kernel_launch(void* const* d_in, const int* in_sizes, int n_in,
                              void* d_out, int out_size, void* d_ws, size_t ws_size,
                              hipStream_t stream) {
  const float* mg = (const float*)d_in[0];
  const float* sg = (const float*)d_in[1];
  const float* Wr = (const float*)d_in[2];
  const float* br = (const float*)d_in[3];
  const float* ga = (const float*)d_in[4];
  const float* be = (const float*)d_in[5];
  const float* Ws = (const float*)d_in[6];
  const float* bs = (const float*)d_in[7];
  const float* Ks = (const float*)d_in[8];
  const float* Ms = (const float*)d_in[9];
  const float* Wt = (const float*)d_in[10];
  const float* bt = (const float*)d_in[11];
  const float* Kt = (const float*)d_in[12];
  const float* Mt = (const float*)d_in[13];
  __bf16* wsb = (__bf16*)d_ws;
  float* out = (float*)d_out;

  moe_prep<<<(WS_TOTAL + 255) / 256, 256, 0, stream>>>(Ws, Wt, Ks, Kt, Ms, Mt, Wr, wsb);
  moe_main<<<250, 32 * WAVES, SMEM_BYTES, stream>>>(mg, sg, br, ga, be, bs, bt, wsb, out);
}
